// MultiHeadAttn_68040871903724
// MI455X (gfx1250) — matrix operand order repaired and checked by reading, not compiled, not run on hardware
//
#include <hip/hip_runtime.h>

typedef __attribute__((ext_vector_type(16))) _Float16 v16h;
typedef __attribute__((ext_vector_type(8)))  _Float16 v8h;
typedef __attribute__((ext_vector_type(4)))  _Float16 v4h;
typedef __attribute__((ext_vector_type(8)))  float    v8f;

#define DEV __device__ __forceinline__

constexpr int cB = 4, cLQ = 2048, cLM = 2048, cHID = 512, cNH = 8, cHD = 64;

union F8  { v8f  v; float f[8]; };
union H16 { v16h v; v8h   h[2]; };

DEV v8f zero_v8f() {
  v8f v;
#pragma unroll
  for (int i = 0; i < 8; ++i) v[i] = 0.0f;
  return v;
}

// ---- CDNA5 async global->LDS copy (ASYNCcnt tracked) ----------------------
DEV void async_b128(const _Float16* gbase, unsigned goff_bytes, unsigned lds_byte_addr) {
  asm volatile("global_load_async_to_lds_b128 %0, %1, %2"
               :: "v"(lds_byte_addr), "v"(goff_bytes), "s"(gbase)
               : "memory");
}
DEV void wait_async0() { asm volatile("s_wait_asynccnt 0" ::: "memory"); }
DEV void wait_ds0()    { asm volatile("s_wait_dscnt 0"    ::: "memory"); }

// A-matrix fragment (16x32 f16, MxK), CDNA5 layout:
// lane<16: row M=lane, K = k0..k0+7 and k0+16..k0+23; lane>=16: k base +8.
DEV v16h ld_frag_a(const _Float16* base, int ld, int lane, int kchunk) {
  int m  = lane & 15;
  int k0 = kchunk * 32 + ((lane >> 4) << 3);
  const _Float16* p = base + (size_t)m * ld + k0;
  H16 u;
  u.h[0] = *(const v8h*)(p);
  u.h[1] = *(const v8h*)(p + 16);
  return u.v;
}

// B-matrix fragment (32x16 f16, KxN) from an N-major buffer:
// lane<16: col N=lane, K = k0..k0+7 and k0+16..k0+23; lane>=16: k base +8.
DEV v16h ld_frag_b(const _Float16* base, size_t ld, int lane, int kchunk) {
  int n  = lane & 15;
  int k0 = kchunk * 32 + ((lane >> 4) << 3);
  const _Float16* p = base + (size_t)n * ld + k0;
  H16 u;
  u.h[0] = *(const v8h*)(p);
  u.h[1] = *(const v8h*)(p + 16);
  return u.v;
}

// Issue async copy of a 64x32-half tile (rows contiguous in source).
DEV void issue_tile_64x32(const _Float16* gsrc, int grow_base, int gld_halves,
                          int kofs_halves, _Float16 (*lds)[40], int tid) {
#pragma unroll
  for (int p = 0; p < 2; ++p) {
    int idx = p * 128 + tid;
    int row = idx >> 2;
    int c8  = (idx & 3) << 3;
    unsigned goff = (unsigned)(((size_t)(grow_base + row) * gld_halves + kofs_halves + c8) * 2);
    async_b128(gsrc, goff, (unsigned)(size_t)&lds[row][c8]);
  }
}

// Issue async copy of a 64x64-half tile.
DEV void issue_tile_64x64(const _Float16* gsrc, int grow_base, int gld_halves,
                          int colofs_halves, _Float16 (*lds)[72], int tid) {
#pragma unroll
  for (int p = 0; p < 4; ++p) {
    int idx = p * 128 + tid;
    int row = idx >> 3;
    int c8  = (idx & 7) << 3;
    unsigned goff = (unsigned)(((size_t)(grow_base + row) * gld_halves + colofs_halves + c8) * 2);
    async_b128(gsrc, goff, (unsigned)(size_t)&lds[row][c8]);
  }
}

// ---------------------------------------------------------------------------
// GEMM: C[M,N] = A[M,K](f16) * Wt[N,K](f16, pre-transposed) + bias (+resid)
// 64x64 tile, 4 waves, double-buffered async global->LDS staging.
// ---------------------------------------------------------------------------
template <int N, int K, bool RESID>
__global__ __launch_bounds__(128) void gemm_async_kernel(
    const _Float16* __restrict__ A, const _Float16* __restrict__ Wt,
    const float* __restrict__ bias, const float* __restrict__ resid,
    float* __restrict__ C)
{
  __shared__ __align__(16) _Float16 As[2][64][40];
  __shared__ __align__(16) _Float16 Bs[2][64][40];
  const int tid = threadIdx.x, lane = tid & 31, wave = tid >> 5;
  const int mbase = blockIdx.y * 64, nbase = blockIdx.x * 64;

  v8f acc[4];
#pragma unroll
  for (int nb = 0; nb < 4; ++nb) acc[nb] = zero_v8f();

  issue_tile_64x32(A,  mbase, K, 0, As[0], tid);
  issue_tile_64x32(Wt, nbase, K, 0, Bs[0], tid);

  constexpr int NT = K / 32;
  for (int kt = 0; kt < NT; ++kt) {
    const int buf = kt & 1;
    wait_async0();
    __syncthreads();
    if (kt + 1 < NT) {
      issue_tile_64x32(A,  mbase, K, (kt + 1) * 32, As[buf ^ 1], tid);
      issue_tile_64x32(Wt, nbase, K, (kt + 1) * 32, Bs[buf ^ 1], tid);
    }
    v16h a = ld_frag_a(&As[buf][wave * 16][0], 40, lane, 0);
    v16h bm[4];
#pragma unroll
    for (int nb = 0; nb < 4; ++nb) bm[nb] = ld_frag_b(&Bs[buf][nb * 16][0], 40, lane, 0);
#pragma unroll
    for (int nb = 0; nb < 4; ++nb)
      acc[nb] = __builtin_amdgcn_wmma_f32_16x16x32_f16(
          false, a, false, bm[nb], (short)0, acc[nb], false, false);
  }

  const int row0 = mbase + wave * 16 + ((lane >> 4) << 3);
  const int col0 = nbase + (lane & 15);
#pragma unroll
  for (int nb = 0; nb < 4; ++nb) {
    F8 u; u.v = acc[nb];
    const int col = col0 + nb * 16;
    const float bv = bias[col];
#pragma unroll
    for (int r = 0; r < 8; ++r) {
      size_t idx = (size_t)(row0 + r) * N + col;
      float v = u.f[r] + bv;
      if constexpr (RESID) v += resid[idx];
      C[idx] = v;
    }
  }
}

// ---------------------------------------------------------------------------
// Prep: f32 -> f16 cast (vectorized) and weight transpose+cast to N-major.
// ---------------------------------------------------------------------------
__global__ __launch_bounds__(256) void cast4_kernel(
    const float* __restrict__ in, _Float16* __restrict__ out, int n4)
{
  int i = blockIdx.x * 256 + threadIdx.x;
  if (i >= n4) return;
  float4 v = ((const float4*)in)[i];
  v4h h;
  h[0] = (_Float16)v.x; h[1] = (_Float16)v.y;
  h[2] = (_Float16)v.z; h[3] = (_Float16)v.w;
  ((v4h*)out)[i] = h;
}

__global__ __launch_bounds__(256) void transpose_cast_kernel(
    const float* __restrict__ W, _Float16* __restrict__ Wt, int K, int N)
{
  int idx = blockIdx.x * 256 + threadIdx.x;
  if (idx >= K * N) return;
  int n = idx % N, k = idx / N;
  Wt[(size_t)n * K + k] = (_Float16)W[(size_t)k * N + n];
}

// ---------------------------------------------------------------------------
// RoPE-style rotation + pack: Qh[B,H,LQ,64], Kh[B,H,LM,64], VtG[B,H,64,LM].
// ---------------------------------------------------------------------------
__global__ __launch_bounds__(256) void rope_pack_kernel(
    const float* __restrict__ Qraw, const float* __restrict__ KVraw,
    const float* __restrict__ psin, const float* __restrict__ pcos,
    const float* __restrict__ ksin, const float* __restrict__ kcos,
    _Float16* __restrict__ Qh, _Float16* __restrict__ Kh,
    _Float16* __restrict__ VtG)
{
  int b   = blockIdx.x / cLQ;
  int pos = blockIdx.x % cLQ;
  size_t rowq = (size_t)b * cLQ + pos;
  const float* q  = Qraw  + rowq * cHID;
  const float* kv = KVraw + rowq * (2 * cHID);
  const float* ps = psin + rowq * (cHD / 2);
  const float* pc = pcos + rowq * (cHD / 2);
  const float* ks = ksin + rowq * (cHD / 2);
  const float* kc = kcos + rowq * (cHD / 2);

#pragma unroll
  for (int it = 0; it < 2; ++it) {
    int c = threadIdx.x + it * 256;
    int h = c >> 6, d = c & 63, dh = d & 31;
    {  // Q rotate
      float s = ps[dh], co = pc[dh];
      float x0 = q[h * 64 + 2 * dh], x1 = q[h * 64 + 2 * dh + 1];
      float val = (d < 32) ? (x0 * co - x1 * s) : (x1 * co + x0 * s);
      Qh[(((size_t)(b * cNH + h)) * cLQ + pos) * cHD + d] = (_Float16)val;
    }
    {  // K rotate
      float s = ks[dh], co = kc[dh];
      float x0 = kv[h * 128 + 2 * dh], x1 = kv[h * 128 + 2 * dh + 1];
      float val = (d < 32) ? (x0 * co - x1 * s) : (x1 * co + x0 * s);
      Kh[(((size_t)(b * cNH + h)) * cLM + pos) * cHD + d] = (_Float16)val;
    }
    {  // V, transposed [d][pos]
      float v = kv[h * 128 + 64 + d];
      VtG[(((size_t)(b * cNH + h)) * cHD + d) * cLM + pos] = (_Float16)v;
    }
  }
}

// ---------------------------------------------------------------------------
// Flash attention: 4 waves per block, 64 query rows (16/wave), full head dim.
// K/V 64-key tiles staged in LDS via async copies, double-buffered so the
// copy of tile j+1 overlaps compute of tile j. Output x written in f16.
// ---------------------------------------------------------------------------
__global__ __launch_bounds__(128) void flash_attn_kernel(
    const _Float16* __restrict__ Qh, const _Float16* __restrict__ Kh,
    const _Float16* __restrict__ VtG, const unsigned char* __restrict__ mask,
    _Float16* __restrict__ Xh)
{
  __shared__ __align__(16) _Float16 Kt[2][64][72];
  __shared__ __align__(16) _Float16 Vs[2][64][72];
  __shared__ __align__(16) _Float16 Plds[4][16][72];
  const int tid = threadIdx.x, lane = tid & 31, wave = tid >> 5;
  const int h = blockIdx.y, b = blockIdx.z;
  const int bh = b * cNH + h;
  const int qbase = blockIdx.x * 64 + wave * 16;
  const _Float16* Qbase = Qh  + ((size_t)bh * cLQ + qbase) * cHD;
  const _Float16* Kbase = Kh  + (size_t)bh * cLM * cHD;
  const _Float16* Vbase = VtG + (size_t)bh * cHD * cLM;
  const unsigned char* mk_row = mask + (size_t)b * cLM;

  v16h qf[2];
#pragma unroll
  for (int kc = 0; kc < 2; ++kc) qf[kc] = ld_frag_a(Qbase, cHD, lane, kc);

  v8f o[4];
#pragma unroll
  for (int nd = 0; nd < 4; ++nd) o[nd] = zero_v8f();
  float mr[8], lr[8];
#pragma unroll
  for (int r = 0; r < 8; ++r) { mr[r] = -__builtin_inff(); lr[r] = 0.f; }

  constexpr int NT = cLM / 64;
  issue_tile_64x64(Kbase, 0, cHD, 0, Kt[0], tid);
  issue_tile_64x64(Vbase, 0, cLM, 0, Vs[0], tid);

  for (int j = 0; j < NT; ++j) {
    const int buf = j & 1;
    const int kb = j * 64;
    wait_async0();
    __syncthreads();
    if (j + 1 < NT) {
      issue_tile_64x64(Kbase, (j + 1) * 64, cHD, 0, Kt[buf ^ 1], tid);
      issue_tile_64x64(Vbase, 0, cLM, (j + 1) * 64, Vs[buf ^ 1], tid);
    }

    // S = (Q * K^T) / 8
    F8 s[4];
#pragma unroll
    for (int nb = 0; nb < 4; ++nb) {
      v16h kf0 = ld_frag_b(&Kt[buf][nb * 16][0], 72, lane, 0);
      v16h kf1 = ld_frag_b(&Kt[buf][nb * 16][0], 72, lane, 1);
      v8f sv = zero_v8f();
      sv = __builtin_amdgcn_wmma_f32_16x16x32_f16(false, qf[0], false, kf0, (short)0, sv, false, false);
      sv = __builtin_amdgcn_wmma_f32_16x16x32_f16(false, qf[1], false, kf1, (short)0, sv, false, false);
      s[nb].v = sv;
    }
#pragma unroll
    for (int nb = 0; nb < 4; ++nb) {
      int key = kb + nb * 16 + (lane & 15);
      bool m = mk_row[key] != 0;
#pragma unroll
      for (int r = 0; r < 8; ++r) {
        float v = s[nb].f[r] * 0.125f;  // 1/sqrt(64)
        s[nb].f[r] = m ? -__builtin_inff() : v;
      }
    }

    // online softmax (row reductions across the 16-lane half: xor 1/2/4/8)
    float alpha[8];
#pragma unroll
    for (int r = 0; r < 8; ++r) {
      float m0 = fmaxf(fmaxf(s[0].f[r], s[1].f[r]), fmaxf(s[2].f[r], s[3].f[r]));
#pragma unroll
      for (int off = 8; off >= 1; off >>= 1) m0 = fmaxf(m0, __shfl_xor(m0, off, 32));
      float mnew = fmaxf(mr[r], m0);
      alpha[r] = (mr[r] == -__builtin_inff()) ? 0.f : __expf(mr[r] - mnew);
      float sum = 0.f;
#pragma unroll
      for (int nb = 0; nb < 4; ++nb) {
        float sv = s[nb].f[r];
        float p = (sv == -__builtin_inff()) ? 0.f : __expf(sv - mnew);
        s[nb].f[r] = p;
        sum += p;
      }
#pragma unroll
      for (int off = 8; off >= 1; off >>= 1) sum += __shfl_xor(sum, off, 32);
      lr[r] = lr[r] * alpha[r] + sum;
      mr[r] = mnew;
    }
#pragma unroll
    for (int nd = 0; nd < 4; ++nd) {
      F8 t; t.v = o[nd];
#pragma unroll
      for (int r = 0; r < 8; ++r) t.f[r] *= alpha[r];
      o[nd] = t.v;
    }

    // P: C-layout -> per-wave LDS region -> A-layout fragments
#pragma unroll
    for (int nb = 0; nb < 4; ++nb) {
      int col = nb * 16 + (lane & 15);
      int rbase = (lane >> 4) << 3;
#pragma unroll
      for (int r = 0; r < 8; ++r)
        Plds[wave][rbase + r][col] = (_Float16)s[nb].f[r];
    }
    wait_ds0();

    v16h pf[2];
#pragma unroll
    for (int kc = 0; kc < 2; ++kc) pf[kc] = ld_frag_a(&Plds[wave][0][0], 72, lane, kc);

    // O += P * V
#pragma unroll
    for (int nd = 0; nd < 4; ++nd) {
      v16h vf0 = ld_frag_b(&Vs[buf][nd * 16][0], 72, lane, 0);
      v16h vf1 = ld_frag_b(&Vs[buf][nd * 16][0], 72, lane, 1);
      o[nd] = __builtin_amdgcn_wmma_f32_16x16x32_f16(false, pf[0], false, vf0, (short)0, o[nd], false, false);
      o[nd] = __builtin_amdgcn_wmma_f32_16x16x32_f16(false, pf[1], false, vf1, (short)0, o[nd], false, false);
    }
  }

  float inv[8];
#pragma unroll
  for (int r = 0; r < 8; ++r) inv[r] = (lr[r] > 0.f) ? 1.f / lr[r] : 0.f;
#pragma unroll
  for (int nd = 0; nd < 4; ++nd) {
    F8 t; t.v = o[nd];
#pragma unroll
    for (int r = 0; r < 8; ++r) {
      int row = qbase + r + ((lane >> 4) << 3);
      int col = h * cHD + nd * 16 + (lane & 15);
      Xh[((size_t)b * cLQ + row) * cHID + col] = (_Float16)(t.f[r] * inv[r]);
    }
  }
}

// ---------------------------------------------------------------------------
// LayerNorm over the 512-wide hidden dim; one block per row.
// ---------------------------------------------------------------------------
__global__ __launch_bounds__(256) void layernorm_kernel(
    const float* __restrict__ Y, const float* __restrict__ gamma,
    const float* __restrict__ beta, float* __restrict__ out)
{
  __shared__ float red[34];
  int row = blockIdx.x;
  const float* y = Y + (size_t)row * cHID;
  int tid = threadIdx.x, lane = tid & 31, wave = tid >> 5;
  float v0 = y[tid], v1 = y[tid + 256];
  float s = v0 + v1, ss = v0 * v0 + v1 * v1;
#pragma unroll
  for (int off = 16; off >= 1; off >>= 1) {
    s  += __shfl_xor(s, off, 32);
    ss += __shfl_xor(ss, off, 32);
  }
  if (lane == 0) { red[wave] = s; red[wave + 8] = ss; }
  __syncthreads();
  if (tid == 0) {
    float ts = 0.f, tss = 0.f;
    for (int i = 0; i < 8; ++i) { ts += red[i]; tss += red[i + 8]; }
    red[32] = ts; red[33] = tss;
  }
  __syncthreads();
  float mu   = red[32] * (1.0f / cHID);
  float var  = red[33] * (1.0f / cHID) - mu * mu;
  float rstd = rsqrtf(var + 1e-5f);
  out[(size_t)row * cHID + tid]       = (v0 - mu) * rstd * gamma[tid]       + beta[tid];
  out[(size_t)row * cHID + tid + 256] = (v1 - mu) * rstd * gamma[tid + 256] + beta[tid + 256];
}

// ---------------------------------------------------------------------------
extern "C" void kernel_launch(void* const* d_in, const int* in_sizes, int n_in,
                              void* d_out, int out_size, void* d_ws, size_t ws_size,
                              hipStream_t stream) {
  (void)in_sizes; (void)n_in; (void)out_size; (void)ws_size;
  const float* tgt   = (const float*)d_in[0];
  const float* mem   = (const float*)d_in[1];
  const unsigned char* maskp = (const unsigned char*)d_in[2];
  const float* psin  = (const float*)d_in[3];
  const float* pcos  = (const float*)d_in[4];
  const float* ksin  = (const float*)d_in[5];
  const float* kcos  = (const float*)d_in[6];
  const float* Wq    = (const float*)d_in[7];
  const float* bq    = (const float*)d_in[8];
  const float* Wkv   = (const float*)d_in[9];
  const float* bkv   = (const float*)d_in[10];
  const float* Wo    = (const float*)d_in[11];
  const float* bo    = (const float*)d_in[12];
  const float* gamma = (const float*)d_in[13];
  const float* beta  = (const float*)d_in[14];

  char* ws = (char*)d_ws;
  size_t off = 0;
  auto take = [&](size_t bytes) -> void* {
    void* p = ws + off;
    off += (bytes + 255) & ~(size_t)255;
    return p;
  };
  _Float16* tgtH  = (_Float16*)take((size_t)cB * cLQ * cHID * 2);
  _Float16* memH  = (_Float16*)take((size_t)cB * cLM * cHID * 2);
  _Float16* WqT   = (_Float16*)take((size_t)cHID * cHID * 2);
  _Float16* WkvT  = (_Float16*)take((size_t)cHID * 2 * cHID * 2);
  _Float16* WoT   = (_Float16*)take((size_t)cHID * cHID * 2);
  float*    Qraw  = (float*)   take((size_t)cB * cLQ * cHID * 4);
  float*    KVraw = (float*)   take((size_t)cB * cLM * 2 * cHID * 4);
  _Float16* Qh    = (_Float16*)take((size_t)cB * cNH * cLQ * cHD * 2);
  _Float16* Kh    = (_Float16*)take((size_t)cB * cNH * cLM * cHD * 2);
  _Float16* VtG   = (_Float16*)take((size_t)cB * cNH * cHD * cLM * 2);
  _Float16* XhBuf = (_Float16*)take((size_t)cB * cLQ * cHID * 2);
  float*    Y     = (float*)   take((size_t)cB * cLQ * cHID * 4);

  const int nAct4 = cB * cLQ * cHID / 4;  // 1048576
  cast4_kernel<<<dim3(nAct4 / 256), dim3(256), 0, stream>>>(tgt, tgtH, nAct4);
  cast4_kernel<<<dim3(nAct4 / 256), dim3(256), 0, stream>>>(mem, memH, nAct4);
  transpose_cast_kernel<<<dim3(cHID * cHID / 256), dim3(256), 0, stream>>>(Wq, WqT, cHID, cHID);
  transpose_cast_kernel<<<dim3(cHID * 2 * cHID / 256), dim3(256), 0, stream>>>(Wkv, WkvT, cHID, 2 * cHID);
  transpose_cast_kernel<<<dim3(cHID * cHID / 256), dim3(256), 0, stream>>>(Wo, WoT, cHID, cHID);

  dim3 blk128(128);
  // Q projection
  gemm_async_kernel<cHID, cHID, false><<<dim3(cHID / 64, (cB * cLQ) / 64), blk128, 0, stream>>>(
      tgtH, WqT, bq, nullptr, Qraw);
  // KV projection
  gemm_async_kernel<2 * cHID, cHID, false><<<dim3((2 * cHID) / 64, (cB * cLM) / 64), blk128, 0, stream>>>(
      memH, WkvT, bkv, nullptr, KVraw);
  // rotation + pack to f16 (V transposed)
  rope_pack_kernel<<<dim3(cB * cLQ), dim3(256), 0, stream>>>(
      Qraw, KVraw, psin, pcos, ksin, kcos, Qh, Kh, VtG);
  // attention -> f16 x
  flash_attn_kernel<<<dim3(cLQ / 64, cNH, cB), blk128, 0, stream>>>(
      Qh, Kh, VtG, maskp, XhBuf);
  // output projection + residual
  gemm_async_kernel<cHID, cHID, true><<<dim3(cHID / 64, (cB * cLQ) / 64), blk128, 0, stream>>>(
      XhBuf, WoT, bo, tgt, Y);
  // layernorm -> output
  layernorm_kernel<<<dim3(cB * cLQ), dim3(256), 0, stream>>>(
      Y, gamma, beta, (float*)d_out);
}
